// EnhancedSiLUAttention_18511309046441
// MI455X (gfx1250) — matrix operand order repaired and checked by reading, not compiled, not run on hardware
//
#include <hip/hip_runtime.h>
#include <cstdint>

// ---------------------------------------------------------------------------
// EnhancedSiLUAttention for MI455X (gfx1250): bf16 WMMA pipeline, f32 accum,
// async-to-LDS tile staging (ASYNCcnt), double-buffered GEMM, fast SiLU.
// B=4, L=2048, D=1024, H=16, d=64.
// ---------------------------------------------------------------------------

typedef __attribute__((ext_vector_type(16))) __bf16 v16bf;
typedef __attribute__((ext_vector_type(8)))  float  v8f;

#define B_  4
#define L_  2048
#define D_  1024
#define H_  16
#define DH_ 64

__device__ __forceinline__ unsigned short f32_to_bf16(float f) {
  unsigned int u = __float_as_uint(f);
  u += 0x7FFFu + ((u >> 16) & 1u);   // round-to-nearest-even
  return (unsigned short)(u >> 16);
}

// SiLU via hardware reciprocal: v_mul + v_exp + v_add + v_rcp + v_mul.
// (Avoids the precise-divide expansion: v_div_scale/v_div_fmas/v_div_fixup.)
__device__ __forceinline__ float silu_fast(float x) {
  return x * __builtin_amdgcn_rcpf(1.0f + __expf(-x));
}

union Frag { v16bf v; unsigned int u[8]; uint4 q[2]; };

// A fragment (16x32 bf16): lanes 0-15 row M=lane hold K=0..15;
// lanes 16-31 same rows hold K=16..31. Source is row-major LDS [row][k].
__device__ __forceinline__ v16bf load_frag_a(const unsigned short* S, int ld,
                                             int row0, int k0, int lane) {
  const int m  = row0 + (lane & 15);
  const int kh = (lane >> 4) << 4;            // 0 or 16
  const unsigned short* base = S + m * ld + k0 + kh;
  Frag f;
#pragma unroll
  for (int p = 0; p < 4; ++p) {
    f.u[p]     = *(const unsigned int*)(base + 2 * p);        // K = kh+2p,+1
    f.u[4 + p] = *(const unsigned int*)(base + 8 + 2 * p);    // K = 8+kh+2p,+1
  }
  return f.v;
}

// B fragment (32x16 bf16): lanes 0-15 col N=lane hold K=0..15; lanes 16-31 hold
// K=16..31. Source is LDS stored [n][k] (k contiguous) with 16B-aligned rows.
__device__ __forceinline__ v16bf load_frag_b(const unsigned short* S, int ld,
                                             int n0, int k0, int lane) {
  const int n  = n0 + (lane & 15);
  const int kh = (lane >> 4) << 4;            // 0 or 16
  const uint4* p = (const uint4*)(S + n * ld + k0 + kh);
  Frag f;
  f.q[0] = p[0];
  f.q[1] = p[1];
  return f.v;
}

// Async copy of 32 contiguous bytes (per lane) global -> LDS. IOFFSET is added
// to BOTH the LDS and global addresses (ISA 10.x async addressing), so a single
// LDS-address VGPR + 64-bit global pointer covers both b128 beats.
__device__ __forceinline__ void async_copy32(unsigned lds_byte_addr,
                                             const void* gptr) {
  asm volatile("global_load_async_to_lds_b128 %0, %1, off\n\t"
               "global_load_async_to_lds_b128 %0, %1, off offset:16"
               :: "v"(lds_byte_addr), "v"(gptr) : "memory");
}

__device__ __forceinline__ void wait_async0() {
  asm volatile("s_wait_asynccnt 0x0" ::: "memory");
}

// Low 32 bits of a flat LDS pointer are the wave-relative LDS byte offset
// (flat LDS address = {SHARED_BASE, lds_offset[31:0]}).
__device__ __forceinline__ unsigned lds_addr(const void* p) {
  return (unsigned)(uintptr_t)p;
}

// ---------------------------------------------------------------------------
// f32 -> bf16 cast
// ---------------------------------------------------------------------------
__global__ void cast_f32_bf16(const float* __restrict__ in,
                              unsigned short* __restrict__ out, int n) {
  int i = blockIdx.x * blockDim.x + threadIdx.x;
  if (i < n) out[i] = f32_to_bf16(in[i]);
}

// ---------------------------------------------------------------------------
// bf16 WMMA GEMM, 128x128x32 tiles, 256 threads = 8 waves (2x4), wave = 64x32.
// Double-buffered LDS; A tiles staged with async-to-LDS, B tiles via registers
// (transpose into [n][k]).
// MODE 0: C = silu(A@B), N=4096; split epilogue -> qk f32 / v bf16 / u f32.
// MODE 1: C = A@B, plain f32 output.
// ---------------------------------------------------------------------------
#define BM 128
#define BN 128
#define BK 32
#define LDS_A 40
#define LDS_B 40

template<int MODE>
__global__ __launch_bounds__(256) void gemm_bf16(
    const unsigned short* __restrict__ A,   // M x K bf16 row-major
    const unsigned short* __restrict__ Bg,  // K x N bf16 row-major
    int M, int N, int K,
    float* __restrict__ out_f,              // MODE0: qk f32 [M][2048]; MODE1: [M][N]
    unsigned short* __restrict__ v_b,       // MODE0: (B,H,L,d) bf16
    float* __restrict__ u_f)                // MODE0: (B,H,L,d) f32
{
  __shared__ unsigned short As[2][BM * LDS_A];
  __shared__ unsigned short Bs[2][BN * LDS_B]; // stored [n][k]

  const int tid  = threadIdx.x;
  const int lane = tid & 31;
  const int w    = tid >> 5;
  const int wm   = w >> 2;                  // 0..1 -> 64 rows
  const int wn   = w & 3;                  // 0..3 -> 32 cols
  const int bm0  = blockIdx.y * BM;
  const int bn0  = blockIdx.x * BN;

  const v8f vzero = {};
  v8f acc[4][2];
#pragma unroll
  for (int i = 0; i < 4; ++i)
#pragma unroll
    for (int j = 0; j < 2; ++j) acc[i][j] = vzero;

  const int arow = tid >> 1, acol = (tid & 1) << 4;   // 128x32, 16 elems/thread
  const int brow = tid >> 3, bcol = (tid & 7) << 4;   // 32x128, 16 elems/thread
  const unsigned short* grA = A + (size_t)(bm0 + arow) * K + acol;

  // ---- preload tile 0 ----
  {
    async_copy32(lds_addr(&As[0][arow * LDS_A + acol]), grA);
    const uint4* bg = (const uint4*)(Bg + (size_t)brow * N + bn0 + bcol);
    uint4 b0 = bg[0], b1 = bg[1];
    unsigned short tmp[16];
    *(uint4*)(&tmp[0]) = b0;
    *(uint4*)(&tmp[8]) = b1;
#pragma unroll
    for (int i = 0; i < 16; ++i)
      Bs[0][(bcol + i) * LDS_B + brow] = tmp[i];
  }
  wait_async0();
  __syncthreads();

  for (int k0 = 0; k0 < K; k0 += BK) {
    const int buf = (k0 >> 5) & 1;
    const int nxt = buf ^ 1;
    const bool more = (k0 + BK) < K;
    uint4 nb0 = {}, nb1 = {};
    if (more) {
      // stage next tile while this one computes
      async_copy32(lds_addr(&As[nxt][arow * LDS_A + acol]), grA + (k0 + BK));
      const uint4* bg =
          (const uint4*)(Bg + (size_t)(k0 + BK + brow) * N + bn0 + bcol);
      nb0 = bg[0];
      nb1 = bg[1];
    }

    v16bf af[4], bfr[2];
#pragma unroll
    for (int i = 0; i < 4; ++i)
      af[i] = load_frag_a(&As[buf][0], LDS_A, wm * 64 + i * 16, 0, lane);
#pragma unroll
    for (int j = 0; j < 2; ++j)
      bfr[j] = load_frag_b(&Bs[buf][0], LDS_B, wn * 32 + j * 16, 0, lane);
#pragma unroll
    for (int i = 0; i < 4; ++i)
#pragma unroll
      for (int j = 0; j < 2; ++j)
        acc[i][j] = __builtin_amdgcn_wmma_f32_16x16x32_bf16(
            false, af[i], false, bfr[j], (short)0, acc[i][j], false, false);

    if (more) {
      unsigned short tmp[16];
      *(uint4*)(&tmp[0]) = nb0;
      *(uint4*)(&tmp[8]) = nb1;
#pragma unroll
      for (int i = 0; i < 16; ++i)
        Bs[nxt][(bcol + i) * LDS_B + brow] = tmp[i];
      wait_async0();
    }
    __syncthreads();
  }

  const int kh = lane >> 4;
  const int nlane = lane & 15;
#pragma unroll
  for (int i = 0; i < 4; ++i) {
#pragma unroll
    for (int j = 0; j < 2; ++j) {
      const int n = bn0 + wn * 32 + j * 16 + nlane;
#pragma unroll
      for (int r = 0; r < 8; ++r) {
        const int m = bm0 + wm * 64 + i * 16 + r + 8 * kh;
        float vv = acc[i][j][r];
        if (MODE == 1) {
          out_f[(size_t)m * N + n] = vv;
        } else {
          vv = silu_fast(vv);
          if (n < 2048) {
            out_f[(size_t)m * 2048 + n] = vv;         // q|k, pre-RoPE f32
          } else {
            const int rem = n & 1023;
            const int hh = rem >> 6, jj = rem & 63;
            const int bb = m >> 11, ll = m & 2047;
            const size_t ho = ((((size_t)bb * H_ + hh) * L_) + ll) * DH_ + jj;
            if (n < 3072) v_b[ho] = f32_to_bf16(vv);  // v bf16 head-major
            else          u_f[ho] = vv;               // u f32 head-major
          }
        }
      }
    }
  }
}

// ---------------------------------------------------------------------------
// RoPE on q,k (f32 in) -> bf16 head-major (B,H,L,d). One thread per (b,l,h,j<32)
// pair (j, j+32).
// ---------------------------------------------------------------------------
__global__ __launch_bounds__(256) void rope_pack(
    const float* __restrict__ qk,           // [B*L][2048] : q | k
    unsigned short* __restrict__ q_b,
    unsigned short* __restrict__ k_b)
{
  const unsigned int idx = blockIdx.x * 256u + threadIdx.x;  // B*L*H*32
  const int j = idx & 31;
  const int h = (idx >> 5) & 15;
  const int l = (idx >> 9) & 2047;
  const int b = idx >> 20;
  const float invf = __powf(10000.0f, -(float)(2 * j) * (1.0f / 64.0f));
  float s, c;
  __sincosf((float)l * invf, &s, &c);
  const size_t row = (size_t)(b * L_ + l) * 2048;
  const int col = h * 64 + j;
  const float q1 = qk[row + col],        q2 = qk[row + col + 32];
  const float k1 = qk[row + 1024 + col], k2 = qk[row + 1024 + col + 32];
  const size_t ho = ((((size_t)b * H_ + h) * L_) + l) * DH_ + j;
  q_b[ho]      = f32_to_bf16(q1 * c - q2 * s);
  q_b[ho + 32] = f32_to_bf16(q2 * c + q1 * s);
  k_b[ho]      = f32_to_bf16(k1 * c - k2 * s);
  k_b[ho + 32] = f32_to_bf16(k2 * c + k1 * s);
}

// ---------------------------------------------------------------------------
// Attention: one WG = 64-query tile of one (b,h). Stream 64-key blocks:
// S = QK^T (WMMA) -> scale+bias+silu+mask -> bf16 P -> O += P V (WMMA).
// Q/K tiles staged with async-to-LDS; next K/V tiles prefetched.
// Tail: LayerNorm(d=64) + u-gate -> bf16 (B,L,D).
// 256 threads = 8 waves: wm=0..3 (16-row groups), wn=0..1 (32-col groups).
// ---------------------------------------------------------------------------
#define AT_LD 72

__global__ __launch_bounds__(256) void attn_bf16(
    const unsigned short* __restrict__ q_b,
    const unsigned short* __restrict__ k_b,
    const unsigned short* __restrict__ v_b,
    const float* __restrict__ u_f,
    const float* __restrict__ attn_mask,    // (B,1,L,L)
    const float* __restrict__ conv_mask,    // (B,L)
    const float* __restrict__ conv_bias,    // (H)
    const float* __restrict__ gamma,        // (d)
    const float* __restrict__ beta,         // (d)
    unsigned short* __restrict__ attn_out)  // (B,L,D) bf16
{
  __shared__ unsigned short Qs[64 * AT_LD];
  __shared__ unsigned short Ks[64 * AT_LD];
  __shared__ unsigned short Vt[64 * AT_LD]; // transposed: [j][key]
  __shared__ unsigned short Ps[64 * AT_LD];
  __shared__ float Os[64 * 64];

  const int tid  = threadIdx.x;
  const int lane = tid & 31;
  const int w    = tid >> 5;
  const int wm   = w & 3;
  const int wn   = w >> 2;
  const int l0 = blockIdx.x * 64;
  const int h  = blockIdx.y;
  const int b  = blockIdx.z;

  const size_t head = (((size_t)b * H_ + h) * L_) * DH_;
  const int trow = tid >> 2, tcol = (tid & 3) << 4;  // 64x64 tile loads

  // async Q tile (waited inside first loop iteration, before 2nd barrier)
  async_copy32(lds_addr(&Qs[trow * AT_LD + tcol]),
               q_b + head + (size_t)(l0 + trow) * DH_ + tcol);

  const v8f vzero = {};
  v8f oacc[2];
  oacc[0] = vzero; oacc[1] = vzero;
  const float cbh  = conv_bias[h];
  const int kh     = lane >> 4;
  const int nlane  = lane & 15;

  for (int kb0 = 0; kb0 < L_; kb0 += 64) {
    __syncthreads();   // everyone done reading Ks/Vt/Ps from previous block
    {
      // K tile: async global -> LDS (no VGPR round trip)
      async_copy32(lds_addr(&Ks[trow * AT_LD + tcol]),
                   k_b + head + (size_t)(kb0 + trow) * DH_ + tcol);
      // V tile: through registers (needs transpose into [j][key])
      const uint4* gv =
          (const uint4*)(v_b + head + (size_t)(kb0 + trow) * DH_ + tcol);
      uint4 v0 = gv[0], v1 = gv[1];
      unsigned short tmp[16];
      *(uint4*)(&tmp[0]) = v0;
      *(uint4*)(&tmp[8]) = v1;
#pragma unroll
      for (int i = 0; i < 16; ++i)
        Vt[(tcol + i) * AT_LD + trow] = tmp[i];      // transpose
    }
    wait_async0();
    __syncthreads();

    // prefetch next key-block K/V lines into cache while this block computes
    if (kb0 + 64 < L_) {
      __builtin_prefetch(k_b + head + (size_t)(kb0 + 64 + trow) * DH_ + tcol, 0, 0);
      __builtin_prefetch(v_b + head + (size_t)(kb0 + 64 + trow) * DH_ + tcol, 0, 0);
    }

    // ---- S tile: rows wm*16..+16, key cols wn*32..+32 ----
    v8f sacc[2];
    sacc[0] = vzero; sacc[1] = vzero;
#pragma unroll
    for (int kc = 0; kc < 64; kc += 32) {
      v16bf aq = load_frag_a(Qs, AT_LD, wm * 16, kc, lane);
#pragma unroll
      for (int ns = 0; ns < 2; ++ns) {
        v16bf bk = load_frag_b(Ks, AT_LD, wn * 32 + ns * 16, kc, lane);
        sacc[ns] = __builtin_amdgcn_wmma_f32_16x16x32_bf16(
            false, aq, false, bk, (short)0, sacc[ns], false, false);
      }
    }
    // scale + conversion bias + SiLU + attn_mask -> bf16 P in LDS
#pragma unroll
    for (int ns = 0; ns < 2; ++ns) {
      const int key = wn * 32 + ns * 16 + nlane;
      const float bias = conv_mask[b * L_ + kb0 + key] * cbh;
#pragma unroll
      for (int r = 0; r < 8; ++r) {
        const int row = wm * 16 + r + 8 * kh;
        float s = sacc[ns][r] * 0.125f + bias;       // 1/sqrt(64)
        s = silu_fast(s);
        s *= attn_mask[((size_t)b * L_ + (l0 + row)) * L_ + kb0 + key];
        Ps[row * AT_LD + key] = f32_to_bf16(s);
      }
    }
    __syncthreads();

    // ---- O += P @ V ----
#pragma unroll
    for (int kc = 0; kc < 64; kc += 32) {
      v16bf ap = load_frag_a(Ps, AT_LD, wm * 16, kc, lane);
#pragma unroll
      for (int ns = 0; ns < 2; ++ns) {
        v16bf bv = load_frag_b(Vt, AT_LD, wn * 32 + ns * 16, kc, lane);
        oacc[ns] = __builtin_amdgcn_wmma_f32_16x16x32_bf16(
            false, ap, false, bv, (short)0, oacc[ns], false, false);
      }
    }
  }

  // spill O accumulators to LDS
#pragma unroll
  for (int ns = 0; ns < 2; ++ns) {
    const int col = wn * 32 + ns * 16 + nlane;
#pragma unroll
    for (int r = 0; r < 8; ++r) {
      const int row = wm * 16 + r + 8 * kh;
      Os[row * 64 + col] = oacc[ns][r];
    }
  }
  __syncthreads();

  // LayerNorm over d=64 + u-gate; one thread per query row
  if (tid < 64) {
    const int row = tid;
    float mean = 0.f;
#pragma unroll
    for (int j = 0; j < 64; ++j) mean += Os[row * 64 + j];
    mean *= (1.0f / 64.0f);
    float var = 0.f;
#pragma unroll
    for (int j = 0; j < 64; ++j) {
      float d = Os[row * 64 + j] - mean;
      var += d * d;
    }
    var *= (1.0f / 64.0f);
    const float inv = rsqrtf(var + 1e-5f);
    const size_t urow = head + (size_t)(l0 + row) * DH_;
    unsigned short* orow = attn_out + ((size_t)(b * L_ + l0 + row)) * D_ + h * DH_;
#pragma unroll
    for (int j = 0; j < 64; ++j) {
      float val = (Os[row * 64 + j] - mean) * inv * gamma[j] + beta[j];
      val *= u_f[urow + j];
      orow[j] = f32_to_bf16(val);
    }
  }
}

// ---------------------------------------------------------------------------
// Launch
// ---------------------------------------------------------------------------
extern "C" void kernel_launch(void* const* d_in, const int* in_sizes, int n_in,
                              void* d_out, int out_size, void* d_ws, size_t ws_size,
                              hipStream_t stream) {
  const float* x     = (const float*)d_in[0];
  const float* amask = (const float*)d_in[1];
  const float* cmask = (const float*)d_in[2];
  const float* wqkuv = (const float*)d_in[3];
  const float* wout  = (const float*)d_in[4];
  const float* gamma = (const float*)d_in[5];
  const float* beta  = (const float*)d_in[6];
  const float* cbias = (const float*)d_in[7];
  float* out = (float*)d_out;

  char* ws = (char*)d_ws;
  unsigned short* xb  = (unsigned short*)(ws);                            // 16 MB
  unsigned short* wqb = (unsigned short*)(ws + (size_t)16  * (1u << 20)); //  8 MB
  unsigned short* wob = (unsigned short*)(ws + (size_t)24  * (1u << 20)); //  2 MB
  float*          qkf = (float*)         (ws + (size_t)26  * (1u << 20)); // 64 MB
  unsigned short* qb  = (unsigned short*)(ws + (size_t)90  * (1u << 20)); // 16 MB
  unsigned short* kb  = (unsigned short*)(ws + (size_t)106 * (1u << 20)); // 16 MB
  unsigned short* vb  = (unsigned short*)(ws + (size_t)122 * (1u << 20)); // 16 MB
  float*          uf  = (float*)         (ws + (size_t)138 * (1u << 20)); // 32 MB
  unsigned short* ab  = (unsigned short*)(ws + (size_t)170 * (1u << 20)); // 16 MB (186 MB)

  // 1) casts to bf16
  cast_f32_bf16<<<8388608 / 256, 256, 0, stream>>>(x,     xb,  8388608);
  cast_f32_bf16<<<4194304 / 256, 256, 0, stream>>>(wqkuv, wqb, 4194304);
  cast_f32_bf16<<<1048576 / 256, 256, 0, stream>>>(wout,  wob, 1048576);

  // 2) qkuv = silu(x @ Wqkuv); split into qk(f32), v(bf16 head-major), u(f32)
  gemm_bf16<0><<<dim3(4096 / BN, 8192 / BM), 256, 0, stream>>>(
      xb, wqb, 8192, 4096, 1024, qkf, vb, uf);

  // 3) RoPE on q,k -> bf16 head-major
  rope_pack<<<4194304 / 256, 256, 0, stream>>>(qkf, qb, kb);

  // 4) streaming SiLU-attention + LayerNorm + u-gate -> bf16 (B,L,D)
  attn_bf16<<<dim3(L_ / 64, H_, B_), 256, 0, stream>>>(
      qb, kb, vb, uf, amask, cmask, cbias, gamma, beta, ab);

  // 5) out = attn @ Wout (f32 output)
  gemm_bf16<1><<<dim3(1024 / BN, 8192 / BM), 256, 0, stream>>>(
      ab, wob, 8192, 1024, 1024, out, nullptr, nullptr);
}
